// CrossFusionBlock_37915971289779
// MI455X (gfx1250) — matrix operand order repaired and checked by reading, not compiled, not run on hardware
//
#include <hip/hip_runtime.h>
#include <hip/hip_bf16.h>

typedef __attribute__((ext_vector_type(16))) _Float16 v16h;
typedef __attribute__((ext_vector_type(8)))  float    v8f;
typedef _Float16 half_t;

typedef unsigned int u32x4 __attribute__((ext_vector_type(4)));
typedef int          i32x8 __attribute__((ext_vector_type(8)));
typedef int          i32x4 __attribute__((ext_vector_type(4)));

#define SCALE_Q 0.17677669529663687f  // 1/sqrt(32)
#define SMEM_BYTES 145408
#define WS_HALVES_PER_DIR 196608      // WqT 65536 + WkT 32768 + WvT 32768 + WoT 65536
#define WS_BYTES_NEEDED   786432

// LDS layout (bytes):
//   [0,1024)         row stats: [2 tensors][64 rows][mean,rstd] f32
//   [1024,34816)     Qm   [64][264] f16
//   [34816,70656)    RA : nq [64][264], later Km[64][136] + Vt[128][72], later delta [64][264]
//   [70656,104448)   RB : nkv [64][264], later AO [64][264]
//   [104448,124928)  panel0 [NOUT<=256][40] f16   (also per-wave P [8][16][72] in attention)
//   [124928,145408)  panel1 [NOUT<=256][40] f16   (TDM double buffer)

struct CFBParams {
  const float* x_spec;
  const float* x_spat;
  const float* ln[8];   // spec_q g,b, spec_kv g,b, spat_q g,b, spat_kv g,b
  const float* w[16];   // s2t: Wq,bq,Wk,bk,Wv,bv,Wo,bo ; t2s same
  const float* gate_spec;
  const float* gate_spat;
  const half_t* wt;     // packed f16 transposed weights in workspace (TDM path)
  float* out;
};

union V16U { uint4 u[2]; v16h v; };
union H8U  { half_t h[8]; uint4 u; };
union H2U  { half_t h[2]; uint32_t u; };

__device__ __forceinline__ v8f wmma16x16x32(v16h a, v16h b, v8f c) {
  return __builtin_amdgcn_wmma_f32_16x16x32_f16(false, a, false, b, (short)0, c,
                                                false, false);
}

// TDM: DMA a K-panel tile  WT[0..nout-1][k0..k0+31]  (row stride 256 halves) into LDS,
// packed with +8-half row padding (pad_interval=16 DWORDs, pad_amount=4 DWORDs -> ld 40).
__device__ __forceinline__ void tdm_load_panel(half_t* lds_dst, const half_t* gsrc, int nout) {
  const unsigned long long ga = (unsigned long long)(uintptr_t)gsrc;
  u32x4 g0;
  g0[0] = 1u;                                              // count=1 (valid), user mode
  g0[1] = (uint32_t)(uintptr_t)lds_dst;                    // lds_addr (bytes, low 32 of flat)
  g0[2] = (uint32_t)ga;                                    // global_addr[31:0]
  g0[3] = (uint32_t)((ga >> 32) & 0x01FFFFFFu) | (2u << 30);  // addr[56:32] | type=2
  i32x8 g1;
  g1[0] = (1 << 16) | (1 << 20) | (3 << 22) | (3 << 25);   // data_size=2B, pad_en, intv=16dw, amt=4dw
  g1[1] = (int)(256u << 16);                               // tensor_dim0 = 256 (k extent)
  g1[2] = (int)((unsigned)nout << 16);                     // tensor_dim1 = nout
  g1[3] = (int)(32u << 16);                                // tile_dim0 = 32 (k)
  g1[4] = nout;                                            // tile_dim1 = nout, tile_dim2 = 0
  g1[5] = 256;                                             // tensor_dim0_stride = 256
  g1[6] = 0;
  g1[7] = 0;
  i32x4 z4 = {0, 0, 0, 0};                                 // 2-D tensor: groups 2/3 unused
  i32x8 z8 = {0, 0, 0, 0, 0, 0, 0, 0};                     // VADDR4: unused, NULL
  __builtin_amdgcn_tensor_load_to_lds(g0, g1, z4, z4, z8, 0);
}

// A operand: 16(M) x 32(K) f16 tile, element (m,k) at m[m*ld + k].
// lane: M=lane&15, hf=lane>>4; VGPR 0-3: K=8hf+0..7, VGPR 4-7: K=16+8hf+0..7
__device__ __forceinline__ v16h lds_load_A(const half_t* m, int ld, int lane) {
  const int Mr = lane & 15;
  const int hf = lane >> 4;
  const half_t* p = m + Mr * ld + 8 * hf;
  V16U t;
  t.u[0] = *(const uint4*)(p);
  t.u[1] = *(const uint4*)(p + 16);
  return t.v;
}

// B operand from K-transposed storage: element (k,n) at m[n*ld + k].
__device__ __forceinline__ v16h lds_load_Bt(const half_t* m, int ld, int lane) {
  const int Nc = lane & 15;
  const int hf = lane >> 4;
  const half_t* p = m + Nc * ld + 8 * hf;
  V16U t;
  t.u[0] = *(const uint4*)(p);
  t.u[1] = *(const uint4*)(p + 16);
  return t.v;
}

// C layout: lane: N=lane&15, hf=lane>>4; VGPR v holds row M = v + 8*hf
__device__ __forceinline__ void lds_store_C_f16(half_t* m, int ld, v8f c, int lane) {
  const int Nc = lane & 15;
  const int hf = lane >> 4;
#pragma unroll
  for (int v = 0; v < 8; ++v) {
    m[(v + 8 * hf) * ld + Nc] = (half_t)c[v];
  }
}

__device__ __forceinline__ void ln_row8(const float* __restrict__ x,
                                        const float* __restrict__ g,
                                        const float* __restrict__ be,
                                        float mu, float rs,
                                        half_t* __restrict__ dst, int e0) {
  float4 x0 = *(const float4*)(x + e0);
  float4 x1 = *(const float4*)(x + e0 + 4);
  float4 g0 = *(const float4*)(g + e0);
  float4 g1 = *(const float4*)(g + e0 + 4);
  float4 b0 = *(const float4*)(be + e0);
  float4 b1 = *(const float4*)(be + e0 + 4);
  H8U pk;
  pk.h[0] = (half_t)((x0.x - mu) * rs * g0.x + b0.x);
  pk.h[1] = (half_t)((x0.y - mu) * rs * g0.y + b0.y);
  pk.h[2] = (half_t)((x0.z - mu) * rs * g0.z + b0.z);
  pk.h[3] = (half_t)((x0.w - mu) * rs * g0.w + b0.w);
  pk.h[4] = (half_t)((x1.x - mu) * rs * g1.x + b1.x);
  pk.h[5] = (half_t)((x1.y - mu) * rs * g1.y + b1.y);
  pk.h[6] = (half_t)((x1.z - mu) * rs * g1.z + b1.z);
  pk.h[7] = (half_t)((x1.w - mu) * rs * g1.w + b1.w);
  *(uint4*)(dst + e0) = pk.u;
}

// GEMM: [64 x 256(f16 LDS, ld 264)] @ [256 x NOUT]
// TDM=true : B-panels DMA'd from pre-packed f16 W^T via tensor_load_to_lds (double-buffered).
// TDM=false: B-panels staged by ALU from f32 weights (transposed store path).
// MODE 0: Q -> outL ld 264, (acc+bias)*SCALE_Q
// MODE 1: K -> outL ld 136, acc+bias
// MODE 3: V -> outL TRANSPOSED [dim][key] ld 72, vectorized b128 store
// MODE 4: delta -> outL ld 264, acc+bias (O-projection; residual applied in Phase F)
template <int NOUT, int MODE, bool TDM>
__device__ void gemm_step(const half_t* __restrict__ inA,
                          const float* __restrict__ Wg,    // f32 weights (fallback)
                          const half_t* __restrict__ WgT,  // f16 transposed weights (TDM)
                          const float* __restrict__ bias,
                          half_t* __restrict__ pan0,
                          half_t* __restrict__ pan1,
                          half_t* __restrict__ outL,
                          int tid) {
  constexpr int NTW = NOUT / 128;
  constexpr int OLD = (MODE == 1) ? 136 : 264;
  constexpr int NI = NOUT / 32;
  const int w = tid >> 5, lane = tid & 31;
  const int Nc = lane & 15, hf = lane >> 4;

  v8f acc[4][NTW];
  v8f zero = {};
#pragma unroll
  for (int mt = 0; mt < 4; ++mt) {
#pragma unroll
    for (int j = 0; j < NTW; ++j) { acc[mt][j] = zero; }
  }

  const int pr = tid >> 4;
  const int q16 = tid & 15;
  for (int kb = 0; kb < 8; ++kb) {
    half_t* cur = (kb & 1) ? pan1 : pan0;
    if (TDM) {
      if (w == 0) {
        if (kb == 0) tdm_load_panel(cur, WgT, NOUT);                      // prologue
        if (kb < 7) {
          tdm_load_panel((kb & 1) ? pan0 : pan1, WgT + (kb + 1) * 32, NOUT);
          __builtin_amdgcn_s_wait_tensorcnt(1);   // cur complete, next still in flight
        } else {
          __builtin_amdgcn_s_wait_tensorcnt(0);   // last panel complete
        }
      }
    } else {
      // stage transposed K-panel: cur[n][k] = W[kb*32 + k][n], f32 -> f16
      const float* srcA = Wg + (size_t)(kb * 32 + 2 * pr) * NOUT;
      const float* srcB = srcA + NOUT;
#pragma unroll
      for (int i = 0; i < NI; ++i) {
        const int n = 2 * q16 + 32 * i;
        float2 a2 = *(const float2*)(srcA + n);
        float2 b2 = *(const float2*)(srcB + n);
        H2U p0, p1;
        p0.h[0] = (half_t)a2.x; p0.h[1] = (half_t)b2.x;
        p1.h[0] = (half_t)a2.y; p1.h[1] = (half_t)b2.y;
        *(uint32_t*)(cur + (size_t)n * 40 + 2 * pr) = p0.u;
        *(uint32_t*)(cur + (size_t)(n + 1) * 40 + 2 * pr) = p1.u;
      }
      if (kb < 7) __builtin_prefetch(srcA + 32 * NOUT, 0, 1);
    }
    __syncthreads();
    v16h btile[NTW];
#pragma unroll
    for (int j = 0; j < NTW; ++j) {
      btile[j] = lds_load_Bt(cur + (w * NTW + j) * 16 * 40, 40, lane);
    }
#pragma unroll
    for (int mt = 0; mt < 4; ++mt) {
      v16h a = lds_load_A(inA + (mt * 16) * 264 + kb * 32, 264, lane);
#pragma unroll
      for (int j = 0; j < NTW; ++j) {
        acc[mt][j] = wmma16x16x32(a, btile[j], acc[mt][j]);
      }
    }
    __syncthreads();
  }
  // epilogue
#pragma unroll
  for (int mt = 0; mt < 4; ++mt) {
#pragma unroll
    for (int j = 0; j < NTW; ++j) {
      const int nt = w * NTW + j;
      const int n = nt * 16 + Nc;
      const float bv = bias[n];
      if (MODE == 3) {
        H8U pk;
#pragma unroll
        for (int v = 0; v < 8; ++v) { pk.h[v] = (half_t)(acc[mt][j][v] + bv); }
        *(uint4*)(outL + n * 72 + mt * 16 + 8 * hf) = pk.u;
      } else {
#pragma unroll
        for (int v = 0; v < 8; ++v) {
          const int row = mt * 16 + v + 8 * hf;
          float val = acc[mt][j][v] + bv;
          if (MODE == 0) {
            outL[row * OLD + n] = (half_t)(val * SCALE_Q);
          } else {
            outL[row * OLD + n] = (half_t)val;
          }
        }
      }
    }
  }
}

// One wave per query head. Q pre-scaled. kv head = h>>1 (G=2).
__device__ void attn_head(const half_t* __restrict__ Qm,
                          const half_t* __restrict__ Km,
                          const half_t* __restrict__ Vt,
                          half_t* __restrict__ Pw,   // per-wave [16][72]
                          half_t* __restrict__ AO,   // [64][264]
                          int w, int lane) {
  const int kvoff = (w >> 1) * 32;
  const int qoff = w * 32;
  for (int rb = 0; rb < 4; ++rb) {
    v16h aq = lds_load_A(Qm + (rb * 16) * 264 + qoff, 264, lane);
    v8f sc[4];
#pragma unroll
    for (int ct = 0; ct < 4; ++ct) {
      v16h bk = lds_load_Bt(Km + (ct * 16) * 136 + kvoff, 136, lane);
      v8f z = {};
      sc[ct] = wmma16x16x32(aq, bk, z);
    }
    v8f mx = sc[0];
#pragma unroll
    for (int ct = 1; ct < 4; ++ct) {
#pragma unroll
      for (int i = 0; i < 8; ++i) { mx[i] = fmaxf(mx[i], sc[ct][i]); }
    }
#pragma unroll
    for (int msk = 1; msk <= 8; msk <<= 1) {
#pragma unroll
      for (int i = 0; i < 8; ++i) { mx[i] = fmaxf(mx[i], __shfl_xor(mx[i], msk, 32)); }
    }
    v8f sm = {};
#pragma unroll
    for (int ct = 0; ct < 4; ++ct) {
#pragma unroll
      for (int i = 0; i < 8; ++i) {
        sc[ct][i] = __expf(sc[ct][i] - mx[i]);
        sm[i] += sc[ct][i];
      }
    }
#pragma unroll
    for (int msk = 1; msk <= 8; msk <<= 1) {
#pragma unroll
      for (int i = 0; i < 8; ++i) { sm[i] += __shfl_xor(sm[i], msk, 32); }
    }
    v8f inv;
#pragma unroll
    for (int i = 0; i < 8; ++i) { inv[i] = __builtin_amdgcn_rcpf(sm[i]); }  // v_rcp_f32
#pragma unroll
    for (int ct = 0; ct < 4; ++ct) {
      v8f p;
#pragma unroll
      for (int i = 0; i < 8; ++i) { p[i] = sc[ct][i] * inv[i]; }
      lds_store_C_f16(Pw + ct * 16, 72, p, lane);
    }
    v16h a0 = lds_load_A(Pw + 0, 72, lane);
    v16h a1 = lds_load_A(Pw + 32, 72, lane);
#pragma unroll
    for (int nt = 0; nt < 2; ++nt) {
      v8f o = {};
      o = wmma16x16x32(a0, lds_load_Bt(Vt + (kvoff + nt * 16) * 72, 72, lane), o);
      o = wmma16x16x32(a1, lds_load_Bt(Vt + (kvoff + nt * 16) * 72 + 32, 72, lane), o);
      lds_store_C_f16(AO + (rb * 16) * 264 + qoff + nt * 16, 264, o, lane);
    }
  }
}

template <bool TDM>
__global__ __launch_bounds__(256) void cfb_kernel(CFBParams P) {
  extern __shared__ char smem[];
  float* stats = (float*)smem;
  half_t* Qm = (half_t*)(smem + 1024);
  half_t* RA = (half_t*)(smem + 34816);
  half_t* RB = (half_t*)(smem + 70656);
  half_t* pan0 = (half_t*)(smem + 104448);
  half_t* pan1 = (half_t*)(smem + 124928);
  half_t* Km = RA;                 // [64][136]
  half_t* Vt = RA + 64 * 136;      // [128][72]

  const int bc = blockIdx.x;
  const int b = bc >> 7, c = bc & 127;
  const int tid = threadIdx.x, w = tid >> 5, lane = tid & 31;
  const int e0 = lane * 8;

  // Phase 0: per-row mean/rstd for both tensors
  for (int t = 0; t < 2; ++t) {
    const float* xp = t ? P.x_spat : P.x_spec;
    for (int i = 0; i < 8; ++i) {
      const int r = w * 8 + i;
      const float* row = xp + ((size_t)((b * 64 + r) * 128 + c)) * 256;
      float4 x0 = *(const float4*)(row + e0);
      float4 x1 = *(const float4*)(row + e0 + 4);
      float s = x0.x + x0.y + x0.z + x0.w + x1.x + x1.y + x1.z + x1.w;
      float s2 = x0.x * x0.x + x0.y * x0.y + x0.z * x0.z + x0.w * x0.w +
                 x1.x * x1.x + x1.y * x1.y + x1.z * x1.z + x1.w * x1.w;
#pragma unroll
      for (int msk = 16; msk >= 1; msk >>= 1) {
        s += __shfl_xor(s, msk, 32);
        s2 += __shfl_xor(s2, msk, 32);
      }
      if (lane == 0) {
        float mu = s * (1.f / 256.f);
        float var = s2 * (1.f / 256.f) - mu * mu;
        stats[(t * 64 + r) * 2 + 0] = mu;
        stats[(t * 64 + r) * 2 + 1] = rsqrtf(var + 1e-5f);
      }
    }
  }
  __syncthreads();

  for (int dir = 0; dir < 2; ++dir) {
    const float* xq  = dir ? P.x_spat : P.x_spec;
    const float* xkv = dir ? P.x_spec : P.x_spat;
    const float* stq  = stats + (dir ? 128 : 0);
    const float* stkv = stats + (dir ? 0 : 128);
    const float* g_q  = dir ? P.ln[4] : P.ln[0];
    const float* b_q  = dir ? P.ln[5] : P.ln[1];
    const float* g_kv = dir ? P.ln[2] : P.ln[6];
    const float* b_kv = dir ? P.ln[3] : P.ln[7];
    const float* const* W = P.w + dir * 8;
    const half_t* WT = P.wt + (size_t)dir * WS_HALVES_PER_DIR;  // WqT,+65536 WkT,+98304 WvT,+131072 WoT
    const float* gate = dir ? P.gate_spat : P.gate_spec;
    float* outp = P.out + (size_t)dir * (8ull * 64 * 128 * 256);
    const float gsig = 1.f / (1.f + __expf(-gate[0]));

    // Phase A: LayerNorm -> f16 tiles. nq -> RA, nkv -> RB
    for (int i = 0; i < 8; ++i) {
      const int r = w * 8 + i;
      const size_t roff = ((size_t)((b * 64 + r) * 128 + c)) * 256;
      ln_row8(xq + roff,  g_q,  b_q,  stq[r * 2],  stq[r * 2 + 1],  RA + r * 264, e0);
      ln_row8(xkv + roff, g_kv, b_kv, stkv[r * 2], stkv[r * 2 + 1], RB + r * 264, e0);
    }
    __syncthreads();

    // Phase B: Q = (nq @ Wq + bq) * scale -> Qm
    gemm_step<256, 0, TDM>(RA, W[0], WT, W[1], pan0, pan1, Qm, tid);
    __syncthreads();
    // Phase C: K,V = nkv @ Wk/Wv + bias -> Km (row-major), Vt (transposed)
    gemm_step<128, 1, TDM>(RB, W[2], WT + 65536, W[3], pan0, pan1, Km, tid);
    __syncthreads();
    gemm_step<128, 3, TDM>(RB, W[4], WT + 98304, W[5], pan0, pan1, Vt, tid);
    __syncthreads();
    // Phase D: attention, wave = head; AO -> RB (panels dead: P tiles overlap pan0)
    attn_head(Qm, Km, Vt, pan0 + w * (16 * 72), RB, w, lane);
    __syncthreads();
    // Phase E: delta = AO @ Wo + bo -> RA (f16, ld 264); Km/Vt are dead
    gemm_step<256, 4, TDM>(RB, W[6], WT + 131072, W[7], pan0, pan1, RA, tid);
    __syncthreads();
    // Phase F: out = x + sigmoid(gate) * delta  (fully vectorized b128 global traffic)
    for (int i = 0; i < 8; ++i) {
      const int r = w * 8 + i;
      const size_t roff = ((size_t)((b * 64 + r) * 128 + c)) * 256;
      H8U pk;
      pk.u = *(const uint4*)(RA + r * 264 + e0);
      float4 x0 = *(const float4*)(xq + roff + e0);
      float4 x1 = *(const float4*)(xq + roff + e0 + 4);
      float4 o0, o1;
      o0.x = x0.x + gsig * (float)pk.h[0];
      o0.y = x0.y + gsig * (float)pk.h[1];
      o0.z = x0.z + gsig * (float)pk.h[2];
      o0.w = x0.w + gsig * (float)pk.h[3];
      o1.x = x1.x + gsig * (float)pk.h[4];
      o1.y = x1.y + gsig * (float)pk.h[5];
      o1.z = x1.z + gsig * (float)pk.h[6];
      o1.w = x1.w + gsig * (float)pk.h[7];
      *(float4*)(outp + roff + e0) = o0;
      *(float4*)(outp + roff + e0 + 4) = o1;
    }
    __syncthreads();
  }
}

// Pre-pass: convert W [K=256][N] f32 -> W^T [N][256] f16 (coalesced writes).
__global__ __launch_bounds__(256) void wt_pack_kernel(const float* __restrict__ src,
                                                      half_t* __restrict__ dst, int N) {
  const int o = blockIdx.x * 256 + threadIdx.x;   // o < N*256
  const int k = o & 255;
  const int n = o >> 8;
  dst[o] = (half_t)src[(size_t)k * N + n];
}

extern "C" void kernel_launch(void* const* d_in, const int* in_sizes, int n_in,
                              void* d_out, int out_size, void* d_ws, size_t ws_size,
                              hipStream_t stream) {
  (void)in_sizes; (void)n_in; (void)out_size;
  CFBParams P;
  P.x_spec = (const float*)d_in[0];
  P.x_spat = (const float*)d_in[1];
  for (int i = 0; i < 8; ++i)  P.ln[i] = (const float*)d_in[2 + i];
  for (int i = 0; i < 16; ++i) P.w[i]  = (const float*)d_in[10 + i];
  P.gate_spec = (const float*)d_in[26];
  P.gate_spat = (const float*)d_in[27];
  P.wt = (const half_t*)d_ws;
  P.out = (float*)d_out;

  const bool use_tdm = (d_ws != nullptr) && (ws_size >= (size_t)WS_BYTES_NEEDED);
  dim3 grid(1024), block(256);
  if (use_tdm) {
    // pack weights f16-transposed into workspace, then run the TDM-fed kernel
    for (int d = 0; d < 2; ++d) {
      half_t* base = (half_t*)d_ws + (size_t)d * WS_HALVES_PER_DIR;
      wt_pack_kernel<<<256, 256, 0, stream>>>((const float*)d_in[10 + 8 * d + 0], base, 256);
      wt_pack_kernel<<<128, 256, 0, stream>>>((const float*)d_in[10 + 8 * d + 2], base + 65536, 128);
      wt_pack_kernel<<<128, 256, 0, stream>>>((const float*)d_in[10 + 8 * d + 4], base + 98304, 128);
      wt_pack_kernel<<<256, 256, 0, stream>>>((const float*)d_in[10 + 8 * d + 6], base + 131072, 256);
    }
    cfb_kernel<true><<<grid, block, SMEM_BYTES, stream>>>(P);
  } else {
    cfb_kernel<false><<<grid, block, SMEM_BYTES, stream>>>(P);
  }
}
